// MolGNN_79482664780197
// MI455X (gfx1250) — compile-verified, matrix operand order checked by reading, not run on hardware
//
#include <hip/hip_runtime.h>
#include <hip/hip_bf16.h>

typedef __attribute__((ext_vector_type(16))) _Float16 h16;
typedef __attribute__((ext_vector_type(8)))  _Float16 h8;
typedef __attribute__((ext_vector_type(8)))  float    v8f;

// ---------------------------------------------------------------------------
// lin0: cur[n,d] = relu(x[n,:16] @ lin0_w[d,:16] + b[d])
// ---------------------------------------------------------------------------
__global__ void k_lin0(const float* __restrict__ x, const float* __restrict__ w,
                       const float* __restrict__ b, float* __restrict__ cur, int N) {
  int idx = blockIdx.x * blockDim.x + threadIdx.x;
  if (idx >= N * 32) return;
  int n = idx >> 5, d = idx & 31;
  const float* xr = x + (size_t)n * 16;
  const float* wr = w + d * 16;
  float acc = b[d];
#pragma unroll
  for (int k = 0; k < 16; ++k) acc += xr[k] * wr[k];
  cur[idx] = fmaxf(acc, 0.f);
}

// ---------------------------------------------------------------------------
// edge MLP stage 1: h1[e,h] = relu(edge_attr[e,:5] @ mlp_w1[h,:5] + b1[h]) (fp16)
// ---------------------------------------------------------------------------
__global__ void k_edge_mlp(const float* __restrict__ ea, const float* __restrict__ w1,
                           const float* __restrict__ b1, _Float16* __restrict__ h1, int E) {
  long long idx = (long long)blockIdx.x * blockDim.x + threadIdx.x;
  if (idx >= (long long)E * 128) return;
  int e = (int)(idx >> 7), h = (int)(idx & 127);
  const float* a = ea + (size_t)e * 5;
  const float* w = w1 + h * 5;
  float acc = b1[h];
#pragma unroll
  for (int k = 0; k < 5; ++k) acc += a[k] * w[k];
  h1[idx] = (_Float16)fmaxf(acc, 0.f);
}

// ---------------------------------------------------------------------------
// Pack B = W2' [4096 x 32] (+ bias GEMM step t=128) into WMMA B-fragment order:
//   flat[((t*2+u)*32 + lane)*16 + j]
//   lane: g = lane>>4, n = u*16 + (lane&15)
//   K_local(j) = j<8 ? g*8+j : 16+g*8+(j-8)   (16-bit B 32x16 layout)
//   t<128:  k = t*32+K, i=k>>7, h=k&127, val = mlp_w2[(i*32+n)*128 + h]
//   t==128: val = mlp_b2[K*32 + n]            (msg += x_e @ reshape(b2,32,32))
// ---------------------------------------------------------------------------
__global__ void k_pack_B(const float* __restrict__ w2, const float* __restrict__ b2,
                         _Float16* __restrict__ Bp) {
  int idx = blockIdx.x * blockDim.x + threadIdx.x;
  if (idx >= 129 * 1024) return;
  int j    = idx & 15;
  int lane = (idx >> 4) & 31;
  int u    = (idx >> 9) & 1;
  int t    = idx >> 10;
  int g = lane >> 4;
  int n = u * 16 + (lane & 15);
  int K = (j < 8) ? (g * 8 + j) : (16 + g * 8 + (j - 8));
  float v;
  if (t < 128) {
    int k = t * 32 + K;
    int i = k >> 7, h = k & 127;
    v = w2[(size_t)(i * 32 + n) * 128 + h];
  } else {
    v = b2[K * 32 + n];
  }
  Bp[idx] = (_Float16)v;
}

// ---------------------------------------------------------------------------
// scatter-mean denominator
// ---------------------------------------------------------------------------
__global__ void k_count(const int* __restrict__ dst, float* __restrict__ cnt, int E) {
  int e = blockIdx.x * blockDim.x + threadIdx.x;
  if (e < E) atomicAdd(&cnt[dst[e]], 1.f);
}
__global__ void k_inv(float* __restrict__ c, int N) {
  int n = blockIdx.x * blockDim.x + threadIdx.x;
  if (n < N) c[n] = 1.f / fmaxf(c[n], 1.f);
}

// 8 async global->LDS b128 per thread = one 32KB chunk per 256-thread block.
// INST_OFFSET is added to BOTH the LDS address (VDST) and the global address.
#define ASYNC8(LDSA, VOFF, SBASE)                                              \
  asm volatile(                                                                \
      "global_load_async_to_lds_b128 %0, %1, %2 offset:0\n\t"                  \
      "global_load_async_to_lds_b128 %0, %1, %2 offset:4096\n\t"               \
      "global_load_async_to_lds_b128 %0, %1, %2 offset:8192\n\t"               \
      "global_load_async_to_lds_b128 %0, %1, %2 offset:12288\n\t"              \
      "global_load_async_to_lds_b128 %0, %1, %2 offset:16384\n\t"              \
      "global_load_async_to_lds_b128 %0, %1, %2 offset:20480\n\t"              \
      "global_load_async_to_lds_b128 %0, %1, %2 offset:24576\n\t"              \
      "global_load_async_to_lds_b128 %0, %1, %2 offset:28672\n\t"              \
      :: "v"(LDSA), "v"(VOFF), "s"(SBASE) : "memory")

// ---------------------------------------------------------------------------
// Fused NNConv message + scatter-add:
//   msg[32-edge tile, 32] = Z[32 x 4128] @ Bp[4128 x 32]   (f16 WMMA, fp32 acc)
//   Z[m, t*32+K] = x[m, t>>2] * h1[m, (t&3)*32 + K] (t<128); Z[m,K]=x[m,K] (t=128)
// One wave per 32-edge tile (two 16-row M-tiles sharing B fragments),
// 8 waves / 256-thread block. B streamed via double-buffered async->LDS chunks.
// ---------------------------------------------------------------------------
__global__ __launch_bounds__(256) void k_msg(
    const float* __restrict__ cur, const _Float16* __restrict__ h1,
    const _Float16* __restrict__ Bp, const int* __restrict__ src,
    const int* __restrict__ dst, float* __restrict__ agg, int E) {
  __shared__ _Float16 sB[2][16 * 1024];  // 2 x 32KB chunks (16 K-steps each)

  const int lane = threadIdx.x & 31;
  const int wave = threadIdx.x >> 5;
  const int m    = lane & 15;   // A row within M-tile (C column)
  const int g    = lane >> 4;   // lane group (K split)
  const long long tile = (long long)blockIdx.x * 8 + wave;
  const long long eA   = tile * 32 + m;        // M-tile A: edges [0,16)
  const long long eB   = eA + 16;              // M-tile B: edges [16,32)
  const bool evA = (eA < (long long)E);
  const bool evB = (eB < (long long)E);

  // ---- kick off async load of chunk 0 ------------------------------------
  const unsigned voff  = (unsigned)threadIdx.x * 16u;
  const unsigned ldsA  = (unsigned)(uintptr_t)(&sB[0][0]) + voff;
  const unsigned ldsB  = (unsigned)(uintptr_t)(&sB[1][0]) + voff;
  ASYNC8(ldsA, voff, Bp);

  // ---- gather x rows (fp32 -> fp16 scalars) ------------------------------
  _Float16 xhA[32], xhB[32];
  {
    if (evA) {
      const float4* xr = (const float4*)(cur + (size_t)src[eA] * 32);
#pragma unroll
      for (int c = 0; c < 8; ++c) {
        float4 v = xr[c];
        xhA[c * 4 + 0] = (_Float16)v.x; xhA[c * 4 + 1] = (_Float16)v.y;
        xhA[c * 4 + 2] = (_Float16)v.z; xhA[c * 4 + 3] = (_Float16)v.w;
      }
    } else {
#pragma unroll
      for (int c = 0; c < 32; ++c) xhA[c] = (_Float16)0.f;
    }
    if (evB) {
      const float4* xr = (const float4*)(cur + (size_t)src[eB] * 32);
#pragma unroll
      for (int c = 0; c < 8; ++c) {
        float4 v = xr[c];
        xhB[c * 4 + 0] = (_Float16)v.x; xhB[c * 4 + 1] = (_Float16)v.y;
        xhB[c * 4 + 2] = (_Float16)v.z; xhB[c * 4 + 3] = (_Float16)v.w;
      }
    } else {
#pragma unroll
      for (int c = 0; c < 32; ++c) xhB[c] = (_Float16)0.f;
    }
  }

  // ---- preload h1 A-fragments (4 K-blocks of 32, per M-tile) -------------
  h16 hfA[4], hfB[4];
#pragma unroll
  for (int blk = 0; blk < 4; ++blk) {
    h8 a0 = {}, a1 = {}, b0 = {}, b1 = {};
    if (evA) {
      const _Float16* base = h1 + (size_t)eA * 128 + blk * 32 + g * 8;
      a0 = *(const h8*)(base);
      a1 = *(const h8*)(base + 16);
    }
    if (evB) {
      const _Float16* base = h1 + (size_t)eB * 128 + blk * 32 + g * 8;
      b0 = *(const h8*)(base);
      b1 = *(const h8*)(base + 16);
    }
#pragma unroll
    for (int j = 0; j < 8; ++j) {
      hfA[blk][j] = a0[j]; hfA[blk][j + 8] = a1[j];
      hfB[blk][j] = b0[j]; hfB[blk][j + 8] = b1[j];
    }
  }

  v8f accA0 = {}, accA1 = {}, accB0 = {}, accB1 = {};

  // ---- main K loop: 8 double-buffered chunks x 16 K-steps ----------------
#pragma unroll
  for (int ch = 0; ch < 8; ++ch) {
    if (ch < 7) {
      // stream next chunk into the other buffer (released by previous barrier)
      ASYNC8((ch & 1) ? ldsA : ldsB, voff, Bp + (size_t)(ch + 1) * 16384);
      asm volatile("s_wait_asynccnt 8" ::: "memory");   // chunk ch landed
    } else {
      asm volatile("s_wait_asynccnt 0" ::: "memory");
    }
    __syncthreads();  // every thread's part of chunk ch is in LDS
    const _Float16* bufc = (ch & 1) ? &sB[1][0] : &sB[0][0];
#pragma unroll
    for (int tl = 0; tl < 16; ++tl) {
      const int t   = ch * 16 + tl;
      const int i   = t >> 2;   // constant x-dim within this K-step
      const int blk = t & 3;    // h1 K-block
      h16 xsA, xsB;
#pragma unroll
      for (int j = 0; j < 16; ++j) { xsA[j] = xhA[i]; xsB[j] = xhB[i]; }
      h16 aA = hfA[blk] * xsA;  // 8x v_pk_mul_f16 each
      h16 aB = hfB[blk] * xsB;
      const h16 b0 = *(const h16*)(bufc + ((size_t)(tl * 2 + 0) * 32 + lane) * 16);
      const h16 b1 = *(const h16*)(bufc + ((size_t)(tl * 2 + 1) * 32 + lane) * 16);
      accA0 = __builtin_amdgcn_wmma_f32_16x16x32_f16(false, aA, false, b0, (short)0, accA0, false, false);
      accA1 = __builtin_amdgcn_wmma_f32_16x16x32_f16(false, aA, false, b1, (short)0, accA1, false, false);
      accB0 = __builtin_amdgcn_wmma_f32_16x16x32_f16(false, aB, false, b0, (short)0, accB0, false, false);
      accB1 = __builtin_amdgcn_wmma_f32_16x16x32_f16(false, aB, false, b1, (short)0, accB1, false, false);
    }
    __syncthreads();  // release this buffer for the chunk after next
  }

  // ---- bias GEMM step (t = 128): A = x directly, B from global (L2-hot) --
  {
    h16 abA, abB;
#pragma unroll
    for (int j = 0; j < 8; ++j) {
      abA[j]     = g ? xhA[8 + j]  : xhA[j];       // K = g*8 + j
      abA[j + 8] = g ? xhA[24 + j] : xhA[16 + j];  // K = 16 + g*8 + j
      abB[j]     = g ? xhB[8 + j]  : xhB[j];
      abB[j + 8] = g ? xhB[24 + j] : xhB[16 + j];
    }
    const _Float16* bb = Bp + (size_t)128 * 1024;
    const h16 b0 = *(const h16*)(bb + ((size_t)0 * 32 + lane) * 16);
    const h16 b1 = *(const h16*)(bb + ((size_t)1 * 32 + lane) * 16);
    accA0 = __builtin_amdgcn_wmma_f32_16x16x32_f16(false, abA, false, b0, (short)0, accA0, false, false);
    accA1 = __builtin_amdgcn_wmma_f32_16x16x32_f16(false, abA, false, b1, (short)0, accA1, false, false);
    accB0 = __builtin_amdgcn_wmma_f32_16x16x32_f16(false, abB, false, b0, (short)0, accB0, false, false);
    accB1 = __builtin_amdgcn_wmma_f32_16x16x32_f16(false, abB, false, b1, (short)0, accB1, false, false);
  }

  // ---- scatter-add into agg (C layout: VGPR r -> M = r + g*8, N = lane&15)
  const int col = lane & 15;
#pragma unroll
  for (int r = 0; r < 8; ++r) {
    const long long ea = tile * 32 + (g * 8 + r);
    if (ea < (long long)E) {
      int d = dst[ea];
      atomicAdd(agg + (size_t)d * 32 + col,      accA0[r]);
      atomicAdd(agg + (size_t)d * 32 + 16 + col, accA1[r]);
    }
    const long long eb = tile * 32 + 16 + (g * 8 + r);
    if (eb < (long long)E) {
      int d = dst[eb];
      atomicAdd(agg + (size_t)d * 32 + col,      accB0[r]);
      atomicAdd(agg + (size_t)d * 32 + 16 + col, accB1[r]);
    }
  }
}

// ---------------------------------------------------------------------------
// Combine (scatter-mean + root + relu) and GRU step. 512 thr = 16 nodes/block.
// Weights staged in LDS transposed for conflict-free column reads.
// ---------------------------------------------------------------------------
__global__ __launch_bounds__(512) void k_update(
    const float* __restrict__ cur, const float* __restrict__ agg,
    const float* __restrict__ inv,
    const float* __restrict__ root_w, const float* __restrict__ conv_b,
    const float* __restrict__ wih, const float* __restrict__ whh,
    const float* __restrict__ bih, const float* __restrict__ bhh,
    float* __restrict__ nxt, int N) {
  __shared__ float sRootT[1024];
  __shared__ float sWihT[3072];
  __shared__ float sWhhT[3072];
  __shared__ float sCb[32], sBih[96], sBhh[96];
  __shared__ float sOut[16][32];
  __shared__ float sM[16][32];
  const int tid = threadIdx.x;
  for (int idx = tid; idx < 1024; idx += 512) {
    int k = idx >> 5, d = idx & 31;
    sRootT[k * 32 + d] = root_w[d * 32 + k];
  }
  for (int idx = tid; idx < 3072; idx += 512) {
    int k = idx & 31, gd = idx >> 5;
    sWihT[k * 96 + gd] = wih[gd * 32 + k];
    sWhhT[k * 96 + gd] = whh[gd * 32 + k];
  }
  if (tid < 32) sCb[tid] = conv_b[tid];
  if (tid < 96) { sBih[tid] = bih[tid]; sBhh[tid] = bhh[tid]; }
  const int ln = tid >> 5, d = tid & 31;
  const long long n = (long long)blockIdx.x * 16 + ln;
  float hv = 0.f;
  if (n < N) hv = cur[n * 32 + d];
  sOut[ln][d] = hv;
  __syncthreads();
  float mval = 0.f;
  if (n < N) {
    float accr = 0.f;
#pragma unroll
    for (int k = 0; k < 32; ++k) accr += sOut[ln][k] * sRootT[k * 32 + d];
    mval = fmaxf(agg[n * 32 + d] * inv[n] + accr + sCb[d], 0.f);
  }
  sM[ln][d] = mval;
  __syncthreads();
  if (n < N) {
    float gir = sBih[d], giz = sBih[32 + d], gin = sBih[64 + d];
    float ghr = sBhh[d], ghz = sBhh[32 + d], ghn = sBhh[64 + d];
#pragma unroll
    for (int k = 0; k < 32; ++k) {
      float mk = sM[ln][k], hk = sOut[ln][k];
      gir += mk * sWihT[k * 96 + d];
      giz += mk * sWihT[k * 96 + 32 + d];
      gin += mk * sWihT[k * 96 + 64 + d];
      ghr += hk * sWhhT[k * 96 + d];
      ghz += hk * sWhhT[k * 96 + 32 + d];
      ghn += hk * sWhhT[k * 96 + 64 + d];
    }
    float r  = 1.f / (1.f + __expf(-(gir + ghr)));
    float z  = 1.f / (1.f + __expf(-(giz + ghz)));
    float nn = tanhf(gin + r * ghn);
    nxt[n * 32 + d] = (1.f - z) * nn + z * hv;
  }
}

// ---------------------------------------------------------------------------
// head: out[n] = relu(cur[n] @ lin1.T + b1) @ lin2.T + b2
// ---------------------------------------------------------------------------
__global__ void k_final(const float* __restrict__ cur, const float* __restrict__ w1,
                        const float* __restrict__ b1, const float* __restrict__ w2,
                        const float* __restrict__ b2, float* __restrict__ out, int N) {
  long long n = (long long)blockIdx.x * blockDim.x + threadIdx.x;
  if (n >= N) return;
  float xr[32];
#pragma unroll
  for (int k = 0; k < 32; ++k) xr[k] = cur[n * 32 + k];
  float acc = b2[0];
#pragma unroll 4
  for (int j = 0; j < 32; ++j) {
    float t = b1[j];
#pragma unroll
    for (int k = 0; k < 32; ++k) t += xr[k] * w1[j * 32 + k];
    acc += fmaxf(t, 0.f) * w2[j];
  }
  out[n] = acc;
}

// ---------------------------------------------------------------------------
extern "C" void kernel_launch(void* const* d_in, const int* in_sizes, int n_in,
                              void* d_out, int out_size, void* d_ws, size_t ws_size,
                              hipStream_t stream) {
  const float* x       = (const float*)d_in[0];
  const int*   ei      = (const int*)d_in[1];
  const float* ea      = (const float*)d_in[2];
  const float* lin0_w  = (const float*)d_in[3];
  const float* lin0_b  = (const float*)d_in[4];
  const float* mlp_w1  = (const float*)d_in[5];
  const float* mlp_b1  = (const float*)d_in[6];
  const float* mlp_w2  = (const float*)d_in[7];
  const float* mlp_b2  = (const float*)d_in[8];
  const float* root_w  = (const float*)d_in[9];
  const float* conv_b  = (const float*)d_in[10];
  const float* wih     = (const float*)d_in[11];
  const float* whh     = (const float*)d_in[12];
  const float* bih     = (const float*)d_in[13];
  const float* bhh     = (const float*)d_in[14];
  const float* lin1_w  = (const float*)d_in[15];
  const float* lin1_b  = (const float*)d_in[16];
  const float* lin2_w  = (const float*)d_in[17];
  const float* lin2_b  = (const float*)d_in[18];

  const int N = in_sizes[0] / 16;
  const int E = in_sizes[2] / 5;
  const int* src = ei;
  const int* dst = ei + E;

  char* ws = (char*)d_ws;
  auto al = [](size_t v) { return (v + 255) & ~(size_t)255; };
  size_t off = 0;
  float* cur = (float*)(ws + off); off = al(off + (size_t)N * 32 * 4);
  float* nxt = (float*)(ws + off); off = al(off + (size_t)N * 32 * 4);
  float* agg = (float*)(ws + off); off = al(off + (size_t)N * 32 * 4);
  float* inv = (float*)(ws + off); off = al(off + (size_t)N * 4);
  _Float16* h1 = (_Float16*)(ws + off); off = al(off + (size_t)E * 128 * 2);
  _Float16* Bp = (_Float16*)(ws + off); off = al(off + (size_t)129 * 1024 * 2);

  // one-time precompute
  hipMemsetAsync(inv, 0, (size_t)N * 4, stream);
  k_lin0<<<(N * 32 + 255) / 256, 256, 0, stream>>>(x, lin0_w, lin0_b, cur, N);
  k_edge_mlp<<<(int)(((long long)E * 128 + 255) / 256), 256, 0, stream>>>(ea, mlp_w1, mlp_b1, h1, E);
  k_pack_B<<<(129 * 1024 + 255) / 256, 256, 0, stream>>>(mlp_w2, mlp_b2, Bp);
  k_count<<<(E + 255) / 256, 256, 0, stream>>>(dst, inv, E);
  k_inv<<<(N + 255) / 256, 256, 0, stream>>>(inv, N);

  float* a = cur;
  float* b = nxt;
  const int tiles = (E + 31) / 32;       // 32 edges per wave now
  const int mblk  = (tiles + 7) / 8;
  for (int it = 0; it < 3; ++it) {
    hipMemsetAsync(agg, 0, (size_t)N * 32 * 4, stream);
    k_msg<<<mblk, 256, 0, stream>>>(a, h1, Bp, src, dst, agg, E);
    k_update<<<(N + 15) / 16, 512, 0, stream>>>(a, agg, inv, root_w, conv_b,
                                                wih, whh, bih, bhh, b, N);
    float* t = a; a = b; b = t;
  }
  k_final<<<(N + 255) / 256, 256, 0, stream>>>(a, lin1_w, lin1_b, lin2_w, lin2_b,
                                               (float*)d_out, N);
}
